// NAggN_3942779978164
// MI455X (gfx1250) — compile-verified, matrix operand order checked by reading, not run on hardware
//
#include <hip/hip_runtime.h>

typedef _Float16 h16 __attribute__((ext_vector_type(16)));
typedef _Float16 h8  __attribute__((ext_vector_type(8)));
typedef float    f8  __attribute__((ext_vector_type(8)));

namespace cfg {
constexpr int NB = 16, NS = 8, CIN = 3, IH = 128, IW = 128;
constexpr int FV = 512, KOUT = 10;
constexpr int KK = 147;     // 3*7*7 reduction
constexpr int KP = 160;     // K padded to multiple of 32
constexpr int OH = 32, OW = 32, NPOS = OH * OW;
constexpr int NIMG = NB * NS;
constexpr int TILEB = 16 * KP * 2;   // one pos-tile of im2col, bytes (5120)
}

// ---------------------------------------------------------------------------
// Kernel 0: repack conv weights (FV, C, 7, 7) fp32 -> f16 [FV][KP], zero-pad K
// ---------------------------------------------------------------------------
__global__ void prep_weights(const float* __restrict__ cw, _Float16* __restrict__ wh) {
    int i = blockIdx.x * 256 + threadIdx.x;
    if (i >= cfg::FV * cfg::KP) return;
    int oc = i / cfg::KP;
    int k  = i - oc * cfg::KP;
    float v = (k < cfg::KK) ? cw[oc * cfg::KK + k] : 0.0f;
    wh[i] = (_Float16)v;
}

// ---------------------------------------------------------------------------
// Kernel 1: materialize f16 im2col for a chunk of images:
// gcol[img_local][pos][KP], SAME padding (pad_lo = 1), zero-padded K tail.
// Pure gather, massively parallel -> keeps index math out of the GEMM loop.
// ---------------------------------------------------------------------------
__global__ void build_im2col(const float* __restrict__ x,
                             _Float16* __restrict__ gcol,
                             int img0, int nimg) {
    using namespace cfg;
    int idx = blockIdx.x * 256 + threadIdx.x;
    int total = nimg * NPOS * KP;
    if (idx >= total) return;
    int il   = idx / (NPOS * KP);
    int rest = idx - il * (NPOS * KP);
    int pos  = rest / KP;
    int k    = rest - pos * KP;
    float v = 0.0f;
    if (k < KK) {
        int oy = pos >> 5, ox = pos & 31;
        int c   = k / 49;
        int rem = k - c * 49;
        int ky  = rem / 7, kx = rem - ky * 7;
        int iy = oy * 4 - 1 + ky;
        int ix = ox * 4 - 1 + kx;
        if (iy >= 0 && iy < IH && ix >= 0 && ix < IW)
            v = x[(((size_t)(img0 + il) * CIN + c) * IH + iy) * IW + ix];
    }
    gcol[idx] = (_Float16)v;
}

// ---------------------------------------------------------------------------
// Kernel 2: GEMM conv + bias + ReLU + global avg pool.
// One block (8 wave32) per image. Wave w owns oc-tiles w*4..w*4+3 with
// A-fragments hoisted to VGPRs. B pos-tiles (5120 B, contiguous in gcol) are
// double-buffered into LDS via GLOBAL_LOAD_ASYNC_TO_LDS_B128 issued by wave 0,
// fenced with s_wait_asynccnt + block barrier.
// ---------------------------------------------------------------------------
__launch_bounds__(256, 1)
__global__ void conv_pool(const _Float16* __restrict__ wh,
                          const _Float16* __restrict__ gcol,
                          const float* __restrict__ cb,
                          float* __restrict__ fvs,
                          int img0) {
    using namespace cfg;
    __shared__ _Float16 btile[2][16 * KP];   // 2 x 5120 B
    __shared__ float partial[FV];

    const int il   = blockIdx.x;             // image local to chunk
    const int img  = img0 + il;
    const int tid  = threadIdx.x;
    const int lane = tid & 31;
    const int wave = tid >> 5;
    const int hi   = (lane & 16) ? 1 : 0;    // lane half
    const int off  = hi ? 8 : 0;             // K sub-offset per ISA A/B lane map
    const int mrow = lane & 15;              // A row (oc) / B col (pos)

    for (int i = tid; i < FV; i += 256) partial[i] = 0.0f;

    // Hoist A fragments (weights) + per-slot bias into VGPRs.
    h16  afrag[4][5];
    float breg[4][8];
#pragma unroll
    for (int t = 0; t < 4; ++t) {
        const int oct = wave * 4 + t;
        const int oc  = oct * 16 + mrow;
#pragma unroll
        for (int s = 0; s < 5; ++s) {
            const _Float16* p = wh + (size_t)oc * KP + s * 32 + off;
            union { h16 v; h8 h[2]; } u;
            u.h[0] = *(const h8*)(p);        // K = base..base+7
            u.h[1] = *(const h8*)(p + 16);   // K = base+16..base+23
            afrag[t][s] = u.v;
        }
#pragma unroll
        for (int r = 0; r < 8; ++r)
            breg[t][r] = cb[oct * 16 + r + hi * 8];
    }

    float acc[4][8];
#pragma unroll
    for (int t = 0; t < 4; ++t)
#pragma unroll
        for (int r = 0; r < 8; ++r) acc[t][r] = 0.0f;

    const _Float16* gimg = gcol + (size_t)il * NPOS * KP;

    // Async copy of one pos-tile (contiguous 5120 B) into LDS buffer `buf`.
    auto issue_copy = [&](int buf, int pt) {
        const char* src = (const char*)(gimg + (size_t)pt * 16 * KP);
        unsigned lds_base = (unsigned)(uintptr_t)(&btile[buf][0]);
#pragma unroll
        for (int i = 0; i < 10; ++i) {             // 10 * 32 lanes * 16 B = 5120 B
            int chunk = i * 32 + lane;
            unsigned long long g = (unsigned long long)(src + chunk * 16);
            unsigned l = lds_base + chunk * 16;
            asm volatile("global_load_async_to_lds_b128 %0, %1, off"
                         :: "v"(l), "v"(g) : "memory");
        }
    };

    if (wave == 0) issue_copy(0, 0);

    for (int pt = 0; pt < NPOS / 16; ++pt) {
        const int cur = pt & 1;
        if (wave == 0)
            asm volatile("s_wait_asynccnt 0x0" ::: "memory");
        __syncthreads();                          // btile[cur] ready for all waves
        if (wave == 0 && pt + 1 < NPOS / 16)
            issue_copy(cur ^ 1, pt + 1);          // prefetch next tile

        // B fragments from LDS (shared by this wave's 4 oc-tiles).
        h16 bfrag[5];
#pragma unroll
        for (int s = 0; s < 5; ++s) {
            const _Float16* p = &btile[cur][mrow * KP + s * 32 + off];
            union { h16 v; h8 h[2]; } u;
            u.h[0] = *(const h8*)(p);
            u.h[1] = *(const h8*)(p + 16);
            bfrag[s] = u.v;
        }

#pragma unroll
        for (int t = 0; t < 4; ++t) {
            f8 c = {0.f, 0.f, 0.f, 0.f, 0.f, 0.f, 0.f, 0.f};
#pragma unroll
            for (int s = 0; s < 5; ++s) {
                c = __builtin_amdgcn_wmma_f32_16x16x32_f16(
                        false, afrag[t][s], false, bfrag[s],
                        (short)0, c, false, false);
            }
#pragma unroll
            for (int r = 0; r < 8; ++r) {
                float v = c[r] + breg[t][r];      // bias before ReLU
                acc[t][r] += (v > 0.0f) ? v : 0.0f;
            }
        }
        __syncthreads();                          // readers done before overwrite
    }

    // Pool reduction: (M, lane-N) partial sums -> per-channel totals.
#pragma unroll
    for (int t = 0; t < 4; ++t) {
        const int oct = wave * 4 + t;
#pragma unroll
        for (int r = 0; r < 8; ++r)
            atomicAdd(&partial[oct * 16 + r + hi * 8], acc[t][r]);
    }
    __syncthreads();

    for (int i = tid; i < FV; i += 256)
        fvs[(size_t)img * FV + i] = partial[i] * (1.0f / (float)NPOS);
}

// ---------------------------------------------------------------------------
// Kernel 3: masked softmax attention over slices + sigmoid projection.
// ---------------------------------------------------------------------------
__global__ void attn_proj(const float* __restrict__ fvs,
                          const int* __restrict__ nslice,
                          const float* __restrict__ q0,
                          const float* __restrict__ pw,
                          const float* __restrict__ pb,
                          float* __restrict__ out) {
    using namespace cfg;
    __shared__ float sc[NS];
    __shared__ float aw[NS];
    __shared__ float r0[FV];
    __shared__ float ok[KOUT];

    const int b   = blockIdx.x;
    const int tid = threadIdx.x;
    if (tid < NS)   sc[tid] = 0.0f;
    if (tid < KOUT) ok[tid] = 0.0f;
    __syncthreads();

    const float* fb = fvs + (size_t)b * NS * FV;

    for (int s = 0; s < NS; ++s) {
        float p = 0.0f;
        for (int d = tid; d < FV; d += 256) p += q0[d] * fb[s * FV + d];
        for (int o = 16; o > 0; o >>= 1) p += __shfl_down(p, o, 32);
        if ((tid & 31) == 0) atomicAdd(&sc[s], p);
    }
    __syncthreads();

    if (tid == 0) {
        int n = nslice[b];
        n = n < 1 ? 1 : (n > NS ? NS : n);
        float m = -1e30f;
        for (int s = 0; s < n; ++s) m = sc[s] > m ? sc[s] : m;
        float den = 0.0f;
        for (int s = 0; s < n; ++s) den += __expf(sc[s] - m);
        for (int s = 0; s < NS; ++s)
            aw[s] = (s < n) ? __expf(sc[s] - m) / den : 0.0f;
    }
    __syncthreads();

    for (int d = tid; d < FV; d += 256) {
        float v = 0.0f;
#pragma unroll
        for (int s = 0; s < NS; ++s) v += aw[s] * fb[s * FV + d];
        r0[d] = v;
    }
    __syncthreads();

    for (int k = 0; k < KOUT; ++k) {
        float p = 0.0f;
        for (int d = tid; d < FV; d += 256) p += r0[d] * pw[k * FV + d];
        for (int o = 16; o > 0; o >>= 1) p += __shfl_down(p, o, 32);
        if ((tid & 31) == 0) atomicAdd(&ok[k], p);
    }
    __syncthreads();

    if (tid < KOUT)
        out[b * KOUT + tid] = 1.0f / (1.0f + __expf(-(ok[tid] + pb[tid])));
}

// ---------------------------------------------------------------------------
extern "C" void kernel_launch(void* const* d_in, const int* in_sizes, int n_in,
                              void* d_out, int out_size, void* d_ws, size_t ws_size,
                              hipStream_t stream) {
    using namespace cfg;
    const float* x      = (const float*)d_in[0];
    const int*   nslice = (const int*)d_in[1];
    const float* cw     = (const float*)d_in[2];
    const float* cb     = (const float*)d_in[3];
    const float* q0     = (const float*)d_in[4];
    const float* pw     = (const float*)d_in[5];
    const float* pb     = (const float*)d_in[6];
    float* out = (float*)d_out;

    const size_t whB  = (size_t)FV * KP * 2;          // 160 KB
    const size_t fvsB = (size_t)NIMG * FV * 4;        // 256 KB
    _Float16* wh   = (_Float16*)d_ws;
    float*    fvs  = (float*)((char*)d_ws + whB);
    _Float16* gcol = (_Float16*)((char*)d_ws + whB + fvsB);

    // Largest image-chunk whose im2col fits the workspace (42 MB for all 128).
    int chunk = 128;
    while (chunk > 1 &&
           whB + fvsB + (size_t)chunk * NPOS * KP * 2 > ws_size)
        chunk >>= 1;

    prep_weights<<<(FV * KP + 255) / 256, 256, 0, stream>>>(cw, wh);

    for (int img0 = 0; img0 < NIMG; img0 += chunk) {
        int n = chunk;
        int elems = n * NPOS * KP;
        build_im2col<<<(elems + 255) / 256, 256, 0, stream>>>(x, gcol, img0, n);
        conv_pool<<<n, 256, 0, stream>>>(wh, gcol, cb, fvs, img0);
    }

    attn_proj<<<NB, 256, 0, stream>>>(fvs, nslice, q0, pw, pb, out);
}
